// PoseDecoderSparse_77884936946317
// MI455X (gfx1250) — matrix operand order repaired and checked by reading, not compiled, not run on hardware
//
#include <hip/hip_runtime.h>
#include <hip/hip_bf16.h>
#include <math.h>

// Problem constants (from reference): B2=8 -> NB=4 pairs, N=2048, C=64, HID=64
#define NB 4
#define NPTS 2048
#define CH 64
#define LD 2049           // couplings is (NB, 2049, 2049)
#define SINK_ITERS 100

typedef __attribute__((ext_vector_type(16))) __bf16 v16bf;
typedef __attribute__((ext_vector_type(8)))  float  v8f;

// ---------------- bf16 split helpers (round-to-nearest-even) ----------------
__device__ inline __bf16 f2bf(float x) {
  union { float f; unsigned u; } a; a.f = x;
  unsigned u = a.u;
  u += 0x7fffu + ((u >> 16) & 1u);
  union { unsigned short s; __bf16 h; } o; o.s = (unsigned short)(u >> 16);
  return o.h;
}
__device__ inline float bf2f(__bf16 h) {
  union { unsigned short s; __bf16 h; } i; i.h = h;
  union { unsigned u; float f; } o; o.u = ((unsigned)i.s) << 16;
  return o.f;
}

// ---------------------------------------------------------------------------
// Kernel 1: couplings[b, m, n] = dot(f0[m], f1[n]) / 8 via WMMA bf16 split.
// One wave per 16x16 output tile, 8 waves per block. K=64 -> 2 k-chunks of 32,
// 3 WMMAs per chunk (hi*hi + hi*lo + lo*hi) for near-fp32 accuracy.
// A layout (16-bit A 16x32): lane L (g=L/16, r=L%16) holds row M=r,
//   element e -> K = (e&7) + 8*g + 16*(e>>3).
// B layout (16-bit B 32x16): lane L holds col N=r, element e -> K = (e&7) + 8*g + 16*(e>>3).
// C layout: VGPR v, lane L -> M = v + 8*g, N = r.
// ---------------------------------------------------------------------------
__global__ __launch_bounds__(256)
void gemm_coup(const float* __restrict__ feats, float* __restrict__ coup) {
  const int b  = blockIdx.z;
  const int mt = (blockIdx.y * 8 + threadIdx.y) * 16;
  const int nt = blockIdx.x * 16;
  const int lane = threadIdx.x;
  const int g = lane >> 4, r = lane & 15;
  const float* f0 = feats + (size_t)(2 * b) * NPTS * CH;
  const float* f1 = feats + (size_t)(2 * b + 1) * NPTS * CH;
  const float* arow = f0 + (size_t)(mt + r) * CH;
  const float* brow = f1 + (size_t)(nt + r) * CH;

  v8f acc = {};
  #pragma unroll
  for (int ks = 0; ks < 2; ++ks) {
    const int k0 = 32 * ks;
    v16bf ahi, alo, bhi, blo;
    #pragma unroll
    for (int e = 0; e < 16; ++e) {
      int ka = k0 + (e & 7) + 8 * g + ((e >> 3) << 4);
      float xa = arow[ka];
      __bf16 ha = f2bf(xa);
      ahi[e] = ha;
      alo[e] = f2bf(xa - bf2f(ha));
      int kb = k0 + (e & 7) + 8 * g + ((e >> 3) << 4);
      float xb = brow[kb];
      __bf16 hb = f2bf(xb);
      bhi[e] = hb;
      blo[e] = f2bf(xb - bf2f(hb));
    }
    acc = __builtin_amdgcn_wmma_f32_16x16x32_bf16(false, ahi, false, bhi, (short)0, acc, false, false);
    acc = __builtin_amdgcn_wmma_f32_16x16x32_bf16(false, ahi, false, blo, (short)0, acc, false, false);
    acc = __builtin_amdgcn_wmma_f32_16x16x32_bf16(false, alo, false, bhi, (short)0, acc, false, false);
  }

  float* cb = coup + (size_t)b * LD * LD;
  #pragma unroll
  for (int v = 0; v < 8; ++v) {
    int m = mt + v + 8 * g;
    int n = nt + r;
    cb[(size_t)m * LD + n] = acc[v] * 0.125f;   // 1/sqrt(64)
  }
}

// Fill dustbin row/column with bin_score.
__global__ void fill_bins(float* __restrict__ coup, const float* __restrict__ bin) {
  float alpha = bin[0];
  int idx = blockIdx.x * 256 + threadIdx.x;
  if (idx >= NB * LD) return;
  int b = idx / LD, j = idx % LD;
  float* cb = coup + (size_t)b * LD * LD;
  cb[(size_t)j * LD + (LD - 1)] = alpha;
  cb[(size_t)(LD - 1) * LD + j] = alpha;
}

__global__ void init_uv(float* __restrict__ u, float* __restrict__ v) {
  int i = blockIdx.x * 256 + threadIdx.x;
  if (i < NB * LD) { u[i] = 0.f; v[i] = 0.f; }
}

// ---------------------------------------------------------------------------
// Sinkhorn u-update: u[m] = log_mu[m] - LSE_n(coup[m,n] + v[n]).
// One wave per row, online logsumexp, shuffle combine. v staged in LDS.
// ---------------------------------------------------------------------------
__global__ __launch_bounds__(256)
void sink_u(const float* __restrict__ coup, const float* __restrict__ vvec,
            float* __restrict__ uvec) {
  __shared__ float vs[LD];
  const int b = blockIdx.y;
  const int tid = threadIdx.y * 32 + threadIdx.x;
  for (int i = tid; i < LD; i += 256) vs[i] = vvec[b * LD + i];
  __syncthreads();
  const int m = blockIdx.x * 8 + threadIdx.y;
  if (m >= LD) return;
  const float* row = coup + (size_t)b * LD * LD + (size_t)m * LD;
  float mx = -3.0e38f, sm = 0.f;
  for (int n = threadIdx.x; n < LD; n += 32) {
    float x = row[n] + vs[n];
    if (x > mx) { sm = sm * __expf(mx - x) + 1.f; mx = x; }
    else        { sm += __expf(x - mx); }
  }
  #pragma unroll
  for (int off = 16; off > 0; off >>= 1) {
    float omx = __shfl_xor(mx, off, 32);
    float osm = __shfl_xor(sm, off, 32);
    float nmx = fmaxf(mx, omx);
    sm = sm * __expf(mx - nmx) + osm * __expf(omx - nmx);
    mx = nmx;
  }
  if (threadIdx.x == 0) {
    float norm = -__logf(4096.0f);
    float lmu = (m == NPTS) ? (__logf(2048.0f) + norm) : norm;
    uvec[b * LD + m] = lmu - (mx + __logf(sm));
  }
}

// Sinkhorn v-update: v[n] = log_nu[n] - LSE_m(coup[m,n] + u[m]).
// 32 columns per block, 8 row-strided partials per column, LDS combine.
__global__ __launch_bounds__(256)
void sink_v(const float* __restrict__ coup, const float* __restrict__ uvec,
            float* __restrict__ vvec) {
  __shared__ float us[LD];
  __shared__ float rmx[8][32], rsm[8][32];
  const int b = blockIdx.y;
  const int tid = threadIdx.y * 32 + threadIdx.x;
  for (int i = tid; i < LD; i += 256) us[i] = uvec[b * LD + i];
  __syncthreads();
  const int n = blockIdx.x * 32 + threadIdx.x;
  float mx = -3.0e38f, sm = 0.f;
  if (n < LD) {
    const float* base = coup + (size_t)b * LD * LD + n;
    for (int m = threadIdx.y; m < LD; m += 8) {
      float x = base[(size_t)m * LD] + us[m];
      if (x > mx) { sm = sm * __expf(mx - x) + 1.f; mx = x; }
      else        { sm += __expf(x - mx); }
    }
  }
  rmx[threadIdx.y][threadIdx.x] = mx;
  rsm[threadIdx.y][threadIdx.x] = sm;
  __syncthreads();
  if (threadIdx.y == 0 && n < LD) {
    float fm = rmx[0][threadIdx.x], fs = rsm[0][threadIdx.x];
    #pragma unroll
    for (int j = 1; j < 8; ++j) {
      float omx = rmx[j][threadIdx.x], osm = rsm[j][threadIdx.x];
      float nmx = fmaxf(fm, omx);
      fs = fs * __expf(fm - nmx) + osm * __expf(omx - nmx);
      fm = nmx;
    }
    float norm = -__logf(4096.0f);
    float lnu = (n == NPTS) ? (__logf(2048.0f) + norm) : norm;
    vvec[b * LD + n] = lnu - (fm + __logf(fs));
  }
}

// ---------------------------------------------------------------------------
// Matching: Zc argmax over rows/cols (u[m]/v[n] constants drop out per-axis
// except the varying one, which we add).
// ---------------------------------------------------------------------------
__global__ __launch_bounds__(256)
void argmax_rows(const float* __restrict__ coup, const float* __restrict__ vvec,
                 int* __restrict__ nn12) {
  __shared__ float vs[NPTS];
  const int b = blockIdx.y;
  const int tid = threadIdx.y * 32 + threadIdx.x;
  for (int i = tid; i < NPTS; i += 256) vs[i] = vvec[b * LD + i];
  __syncthreads();
  const int m = blockIdx.x * 8 + threadIdx.y;
  const float* row = coup + (size_t)b * LD * LD + (size_t)m * LD;
  float best = -3.0e38f; int bi = 0;
  for (int n = threadIdx.x; n < NPTS; n += 32) {
    float x = row[n] + vs[n];
    if (x > best || (x == best && n < bi)) { best = x; bi = n; }
  }
  #pragma unroll
  for (int off = 16; off > 0; off >>= 1) {
    float ob = __shfl_xor(best, off, 32);
    int   oi = __shfl_xor(bi, off, 32);
    if (ob > best || (ob == best && oi < bi)) { best = ob; bi = oi; }
  }
  if (threadIdx.x == 0) nn12[b * NPTS + m] = bi;
}

__global__ __launch_bounds__(256)
void argmax_cols(const float* __restrict__ coup, const float* __restrict__ uvec,
                 int* __restrict__ nn21) {
  __shared__ float us[NPTS];
  __shared__ float rb[8][32]; __shared__ int ri[8][32];
  const int b = blockIdx.y;
  const int tid = threadIdx.y * 32 + threadIdx.x;
  for (int i = tid; i < NPTS; i += 256) us[i] = uvec[b * LD + i];
  __syncthreads();
  const int n = blockIdx.x * 32 + threadIdx.x;
  const float* base = coup + (size_t)b * LD * LD + n;
  float best = -3.0e38f; int bi = 0;
  for (int m = threadIdx.y; m < NPTS; m += 8) {
    float x = base[(size_t)m * LD] + us[m];
    if (x > best || (x == best && m < bi)) { best = x; bi = m; }
  }
  rb[threadIdx.y][threadIdx.x] = best; ri[threadIdx.y][threadIdx.x] = bi;
  __syncthreads();
  if (threadIdx.y == 0) {
    #pragma unroll
    for (int j = 1; j < 8; ++j) {
      float ob = rb[j][threadIdx.x]; int oi = ri[j][threadIdx.x];
      if (ob > best || (ob == best && oi < bi)) { best = ob; bi = oi; }
    }
    nn21[b * NPTS + n] = bi;
  }
}

__global__ void match_kernel(const int* __restrict__ nn12, const int* __restrict__ nn21,
                             int* __restrict__ matchv) {
  int idx = blockIdx.x * 256 + threadIdx.x;
  if (idx >= NB * NPTS) return;
  int b = idx / NPTS, n = idx % NPTS;
  int m = nn21[idx];
  bool mutual = (nn12[b * NPTS + m] == n);
  matchv[idx] = mutual ? m : -1;
}

__global__ void kinv_kernel(const float* __restrict__ K, float* __restrict__ kinv) {
  int b = threadIdx.x;
  if (b >= NB) return;
  for (int s = 0; s < 2; ++s) {
    const float* k = K + (b * 2 + s) * 9;
    float a = k[0], bb = k[1], c = k[2], d = k[3], e = k[4], f = k[5],
          g = k[6], h = k[7], i = k[8];
    float A = e * i - f * h, B = -(d * i - f * g), C = d * h - e * g;
    float det = a * A + bb * B + c * C;
    float id = 1.f / det;
    float* o = kinv + (b * 2 + s) * 9;
    o[0] = A * id;              o[1] = -(bb * i - c * h) * id; o[2] = (bb * f - c * e) * id;
    o[3] = B * id;              o[4] = (a * i - c * g) * id;   o[5] = -(a * f - c * d) * id;
    o[6] = C * id;              o[7] = -(a * h - bb * g) * id; o[8] = (a * e - bb * d) * id;
  }
}

// ---------------------------------------------------------------------------
// Confidence MLP: 131 -> 64 -> 64 -> 1, weights in LDS, one thread per point.
// Gathers matched features / scores / msel on the fly.
// ---------------------------------------------------------------------------
__global__ __launch_bounds__(256)
void conf_kernel(const float* __restrict__ feats, const float* __restrict__ scores,
                 const float* __restrict__ coup, const float* __restrict__ uvec,
                 const float* __restrict__ vvec, const int* __restrict__ matchv,
                 const float* __restrict__ W1, const float* __restrict__ b1,
                 const float* __restrict__ W2, const float* __restrict__ b2,
                 const float* __restrict__ W3, const float* __restrict__ b3,
                 float* __restrict__ conf) {
  __shared__ float W1s[64 * 131];
  __shared__ float W2s[64 * 64];
  __shared__ float b1s[64], b2s[64], W3s[64];
  __shared__ float b3s;
  const int tid = threadIdx.x;
  for (int i = tid; i < 64 * 131; i += 256) W1s[i] = W1[i];
  for (int i = tid; i < 64 * 64; i += 256) W2s[i] = W2[i];
  if (tid < 64) { b1s[tid] = b1[tid]; b2s[tid] = b2[tid]; W3s[tid] = W3[tid]; }
  if (tid == 0) b3s = b3[0];
  __syncthreads();

  const int b = blockIdx.y;
  const int i = blockIdx.x * 256 + tid;
  const int mm = matchv[b * NPTS + i];
  const bool valid = mm >= 0;
  const int ms = valid ? mm : 0;
  const float vf = valid ? 1.f : 0.f;
  const float* f0r = feats + ((size_t)(2 * b) * NPTS + i) * CH;
  const float* f1r = feats + ((size_t)(2 * b + 1) * NPTS + ms) * CH;
  const float norm = -__logf(4096.0f);
  const float msel = (coup[(size_t)b * LD * LD + (size_t)i * LD + ms] +
                      uvec[b * LD + i] + vvec[b * LD + ms] - norm) * vf;
  const float sc0 = scores[(2 * b) * NPTS + i];
  const float sc1 = scores[(2 * b + 1) * NPTS + ms] * vf;

  float h1[64];
  #pragma unroll
  for (int o = 0; o < 64; ++o) h1[o] = b1s[o];
  for (int c = 0; c < 131; ++c) {
    float xc;
    if (c < 64)       xc = f0r[c];
    else if (c < 128) xc = f1r[c - 64] * vf;
    else if (c == 128) xc = sc0;
    else if (c == 129) xc = sc1;
    else               xc = msel;
    #pragma unroll
    for (int o = 0; o < 64; ++o) h1[o] = fmaf(W1s[o * 131 + c], xc, h1[o]);
  }
  #pragma unroll
  for (int o = 0; o < 64; ++o) h1[o] = fmaxf(h1[o], 0.f);
  float h2[64];
  #pragma unroll
  for (int o = 0; o < 64; ++o) h2[o] = b2s[o];
  for (int c = 0; c < 64; ++c) {
    float xc = h1[c];
    #pragma unroll
    for (int o = 0; o < 64; ++o) h2[o] = fmaf(W2s[o * 64 + c], xc, h2[o]);
  }
  float z = b3s;
  #pragma unroll
  for (int c = 0; c < 64; ++c) z += W3s[c] * fmaxf(h2[c], 0.f);
  float cf = 1.f / (1.f + __expf(-z));
  conf[b * NPTS + i] = cf * vf;
}

__global__ __launch_bounds__(256)
void wsum_kernel(const float* __restrict__ conf, float* __restrict__ wsum) {
  __shared__ float red[256];
  const int b = blockIdx.x;
  float s = 0.f;
  for (int i = threadIdx.x; i < NPTS; i += 256) s += conf[b * NPTS + i];
  red[threadIdx.x] = s; __syncthreads();
  for (int st = 128; st > 0; st >>= 1) {
    if (threadIdx.x < st) red[threadIdx.x] += red[threadIdx.x + st];
    __syncthreads();
  }
  if (threadIdx.x == 0) wsum[b] = red[0];
}

__device__ inline void bearing(const float* Ki, float px, float py, float* x) {
  x[0] = Ki[0] * px + Ki[1] * py + Ki[2];
  x[1] = Ki[3] * px + Ki[4] * py + Ki[5];
  x[2] = Ki[6] * px + Ki[7] * py + Ki[8];
}

// ---------------------------------------------------------------------------
// M9 = sum_n w_n a a^T, a = vec(x1 x0^T) (9). Upper-tri 45 accumulators.
// ---------------------------------------------------------------------------
__global__ __launch_bounds__(256)
void m9_kernel(const float* __restrict__ kpts, const int* __restrict__ matchv,
               const float* __restrict__ conf, const float* __restrict__ wsum,
               const float* __restrict__ kinv, float* __restrict__ M9) {
  const int b = blockIdx.x;
  const float* Ki0 = kinv + b * 18;
  const float* Ki1 = Ki0 + 9;
  const float wn = 1.f / (wsum[b] + 1e-8f);
  float acc[45];
  #pragma unroll
  for (int k = 0; k < 45; ++k) acc[k] = 0.f;

  for (int i = threadIdx.x; i < NPTS; i += 256) {
    float wgt = conf[b * NPTS + i] * wn;
    const float* kp0 = kpts + ((size_t)(2 * b) * NPTS + i) * 2;
    int mm = matchv[b * NPTS + i];
    int ms = mm >= 0 ? mm : 0;
    float vf = mm >= 0 ? 1.f : 0.f;
    const float* kp1 = kpts + ((size_t)(2 * b + 1) * NPTS + ms) * 2;
    float x0[3], x1[3];
    bearing(Ki0, kp0[0], kp0[1], x0);
    bearing(Ki1, kp1[0] * vf, kp1[1] * vf, x1);
    float a[9];
    #pragma unroll
    for (int ii = 0; ii < 3; ++ii)
      #pragma unroll
      for (int jj = 0; jj < 3; ++jj) a[3 * ii + jj] = x1[ii] * x0[jj];
    int k45 = 0;
    #pragma unroll
    for (int p = 0; p < 9; ++p)
      #pragma unroll
      for (int q = p; q < 9; ++q) { acc[k45] = fmaf(wgt * a[p], a[q], acc[k45]); ++k45; }
  }
  #pragma unroll
  for (int k = 0; k < 45; ++k)
    #pragma unroll
    for (int off = 16; off > 0; off >>= 1) acc[k] += __shfl_xor(acc[k], off, 32);

  __shared__ float part[8][45];
  const int wv = threadIdx.x >> 5, lane = threadIdx.x & 31;
  if (lane == 0) for (int k = 0; k < 45; ++k) part[wv][k] = acc[k];
  __syncthreads();
  if (threadIdx.x == 0) {
    float* out = M9 + b * 81;
    int k45 = 0;
    for (int p = 0; p < 9; ++p)
      for (int q = p; q < 9; ++q) {
        float t = 0.f;
        for (int j = 0; j < 8; ++j) t += part[j][k45];
        out[p * 9 + q] = t; out[q * 9 + p] = t;
        ++k45;
      }
  }
}

// ---------------------------------------------------------------------------
// Per-batch decomposition (4 batches, 1 thread each): 9x9 Jacobi -> smallest
// eigenvector -> E; 3x3 SVD via Jacobi on E^T E; R1, R2, t.
// ---------------------------------------------------------------------------
__device__ void jacobi_sym(float* A, float* V, int n, int sweeps) {
  for (int i = 0; i < n * n; ++i) V[i] = 0.f;
  for (int i = 0; i < n; ++i) V[i * n + i] = 1.f;
  for (int s = 0; s < sweeps; ++s) {
    for (int p = 0; p < n - 1; ++p) {
      for (int q = p + 1; q < n; ++q) {
        float apq = A[p * n + q];
        if (fabsf(apq) < 1e-18f) continue;
        float app = A[p * n + p], aqq = A[q * n + q];
        float theta = 0.5f * (aqq - app) / apq;
        float t = (theta >= 0.f ? 1.f : -1.f) / (fabsf(theta) + sqrtf(theta * theta + 1.f));
        float c = 1.f / sqrtf(t * t + 1.f), sn = t * c;
        for (int k = 0; k < n; ++k) {
          float akp = A[k * n + p], akq = A[k * n + q];
          A[k * n + p] = c * akp - sn * akq;
          A[k * n + q] = sn * akp + c * akq;
        }
        for (int k = 0; k < n; ++k) {
          float apk = A[p * n + k], aqk = A[q * n + k];
          A[p * n + k] = c * apk - sn * aqk;
          A[q * n + k] = sn * apk + c * aqk;
        }
        for (int k = 0; k < n; ++k) {
          float vkp = V[k * n + p], vkq = V[k * n + q];
          V[k * n + p] = c * vkp - sn * vkq;
          V[k * n + q] = sn * vkp + c * vkq;
        }
      }
    }
  }
}
__device__ inline void cross3(const float* a, const float* b, float* c) {
  c[0] = a[1] * b[2] - a[2] * b[1];
  c[1] = a[2] * b[0] - a[0] * b[2];
  c[2] = a[0] * b[1] - a[1] * b[0];
}
__device__ inline float det3(const float* m) {
  return m[0] * (m[4] * m[8] - m[5] * m[7]) - m[1] * (m[3] * m[8] - m[5] * m[6]) +
         m[2] * (m[3] * m[7] - m[4] * m[6]);
}
__device__ inline void matmul3(const float* a, const float* b, float* c) {
  for (int i = 0; i < 3; ++i)
    for (int j = 0; j < 3; ++j) {
      float s = 0.f;
      for (int k = 0; k < 3; ++k) s += a[i * 3 + k] * b[k * 3 + j];
      c[i * 3 + j] = s;
    }
}

__global__ void eig_kernel(const float* __restrict__ M9, float* __restrict__ RT) {
  const int b = blockIdx.x;
  float A[81], V[81];
  for (int i = 0; i < 81; ++i) A[i] = M9[b * 81 + i];
  jacobi_sym(A, V, 9, 30);
  int idx = 0; float mn = A[0];
  for (int k = 1; k < 9; ++k) { float d = A[k * 9 + k]; if (d < mn) { mn = d; idx = k; } }
  float E[9];
  for (int r = 0; r < 9; ++r) E[r] = V[r * 9 + idx];

  // SVD(E) via eigendecomp of E^T E
  float A3[9], V3[9];
  for (int i = 0; i < 3; ++i)
    for (int j = 0; j < 3; ++j) {
      float s = 0.f;
      for (int k = 0; k < 3; ++k) s += E[k * 3 + i] * E[k * 3 + j];
      A3[i * 3 + j] = s;
    }
  jacobi_sym(A3, V3, 3, 15);
  float ev[3] = { A3[0], A3[4], A3[8] };
  int ord[3] = { 0, 1, 2 };
  for (int a1 = 0; a1 < 2; ++a1)
    for (int a2 = a1 + 1; a2 < 3; ++a2)
      if (ev[ord[a2]] > ev[ord[a1]]) { int t = ord[a1]; ord[a1] = ord[a2]; ord[a2] = t; }
  float v0[3], v1[3], v2[3];
  for (int r = 0; r < 3; ++r) {
    v0[r] = V3[r * 3 + ord[0]]; v1[r] = V3[r * 3 + ord[1]]; v2[r] = V3[r * 3 + ord[2]];
  }
  float s0 = sqrtf(fmaxf(ev[ord[0]], 0.f)), s1 = sqrtf(fmaxf(ev[ord[1]], 0.f));
  float u0[3], u1[3], u2[3];
  for (int r = 0; r < 3; ++r) {
    u0[r] = (E[r * 3 + 0] * v0[0] + E[r * 3 + 1] * v0[1] + E[r * 3 + 2] * v0[2]) / fmaxf(s0, 1e-12f);
    u1[r] = (E[r * 3 + 0] * v1[0] + E[r * 3 + 1] * v1[1] + E[r * 3 + 2] * v1[2]) / fmaxf(s1, 1e-12f);
  }
  cross3(u0, u1, u2);
  float n2 = fmaxf(sqrtf(u2[0] * u2[0] + u2[1] * u2[1] + u2[2] * u2[2]), 1e-12f);
  u2[0] /= n2; u2[1] /= n2; u2[2] /= n2;
  float U[9]  = { u0[0], u1[0], u2[0],  u0[1], u1[1], u2[1],  u0[2], u1[2], u2[2] };
  float Vt[9] = { v0[0], v0[1], v0[2],  v1[0], v1[1], v1[2],  v2[0], v2[1], v2[2] };
  float sU = det3(U)  >= 0.f ? 1.f : -1.f;
  float sV = det3(Vt) >= 0.f ? 1.f : -1.f;
  for (int i = 0; i < 9; ++i) { U[i] *= sU; Vt[i] *= sV; }
  const float Wm[9]  = { 0, -1, 0,  1, 0, 0,  0, 0, 1 };
  const float WmT[9] = { 0, 1, 0,  -1, 0, 0,  0, 0, 1 };
  float UW[9], R1[9], R2[9];
  matmul3(U, Wm, UW);  matmul3(UW, Vt, R1);
  matmul3(U, WmT, UW); matmul3(UW, Vt, R2);
  float* o = RT + b * 21;
  for (int i = 0; i < 9; ++i) { o[i] = R1[i]; o[9 + i] = R2[i]; }
  o[18] = U[2]; o[19] = U[5]; o[20] = U[8];   // t = U[:,2]
}

// Cheirality check over 4 hypotheses (R1,t),(R1,-t),(R2,t),(R2,-t).
__global__ __launch_bounds__(256)
void cheir_kernel(const float* __restrict__ kpts, const int* __restrict__ matchv,
                  const float* __restrict__ conf, const float* __restrict__ wsum,
                  const float* __restrict__ kinv, const float* __restrict__ RT,
                  float* __restrict__ score4) {
  const int b = blockIdx.x;
  const float* R1 = RT + b * 21;
  const float* R2 = RT + b * 21 + 9;
  const float* tv = RT + b * 21 + 18;
  const float* Ki0 = kinv + b * 18;
  const float* Ki1 = Ki0 + 9;
  const float wn = 1.f / (wsum[b] + 1e-8f);
  float sc[4] = { 0.f, 0.f, 0.f, 0.f };

  for (int i = threadIdx.x; i < NPTS; i += 256) {
    float wgt = conf[b * NPTS + i] * wn;
    const float* kp0 = kpts + ((size_t)(2 * b) * NPTS + i) * 2;
    int mm = matchv[b * NPTS + i];
    int ms = mm >= 0 ? mm : 0;
    float vf = mm >= 0 ? 1.f : 0.f;
    const float* kp1 = kpts + ((size_t)(2 * b + 1) * NPTS + ms) * 2;
    float x0[3], x1[3];
    bearing(Ki0, kp0[0], kp0[1], x0);
    bearing(Ki1, kp1[0] * vf, kp1[1] * vf, x1);
    #pragma unroll
    for (int k = 0; k < 4; ++k) {
      const float* R = (k < 2) ? R1 : R2;
      float tsgn = (k & 1) ? -1.f : 1.f;
      float tk[3] = { tv[0] * tsgn, tv[1] * tsgn, tv[2] * tsgn };
      float Rx[3];
      Rx[0] = R[0] * x0[0] + R[1] * x0[1] + R[2] * x0[2];
      Rx[1] = R[3] * x0[0] + R[4] * x0[1] + R[5] * x0[2];
      Rx[2] = R[6] * x0[0] + R[7] * x0[1] + R[8] * x0[2];
      float c[3], ct[3];
      cross3(x1, Rx, c);
      cross3(x1, tk, ct);
      float num = -(ct[0] * c[0] + ct[1] * c[1] + ct[2] * c[2]);
      float den = c[0] * c[0] + c[1] * c[1] + c[2] * c[2] + 1e-9f;
      float d0 = num / den;
      float d1 = d0 * Rx[2] + tk[2];
      if (d0 > 0.f && d1 > 0.f) sc[k] += wgt;
    }
  }
  __shared__ float red[256];
  for (int k = 0; k < 4; ++k) {
    red[threadIdx.x] = sc[k]; __syncthreads();
    for (int st = 128; st > 0; st >>= 1) {
      if (threadIdx.x < st) red[threadIdx.x] += red[threadIdx.x + st];
      __syncthreads();
    }
    if (threadIdx.x == 0) score4[b * 4 + k] = red[0];
    __syncthreads();
  }
}

// Pick best hypothesis, build T and T^{-1}; output (NB, 2, 4, 4).
__global__ void final_kernel(const float* __restrict__ RT, const float* __restrict__ score4,
                             const float* __restrict__ t_scale, float* __restrict__ out) {
  const int b = blockIdx.x;
  const float* s = score4 + b * 4;
  int best = 0; float bv = s[0];
  for (int k = 1; k < 4; ++k) if (s[k] > bv) { bv = s[k]; best = k; }
  const float* R = RT + b * 21 + ((best < 2) ? 0 : 9);
  float tsgn = (best & 1) ? -1.f : 1.f;
  float t[3] = { RT[b * 21 + 18] * tsgn, RT[b * 21 + 19] * tsgn, RT[b * 21 + 20] * tsgn };
  float nrm = sqrtf(t[0] * t[0] + t[1] * t[1] + t[2] * t[2]);
  float scl = t_scale[b] / (nrm + 1e-9f);
  t[0] *= scl; t[1] *= scl; t[2] *= scl;
  float* T  = out + b * 32;
  float* Ti = out + b * 32 + 16;
  for (int i = 0; i < 3; ++i) {
    for (int j = 0; j < 3; ++j) T[i * 4 + j] = R[i * 3 + j];
    T[i * 4 + 3] = t[i];
  }
  T[12] = 0.f; T[13] = 0.f; T[14] = 0.f; T[15] = 1.f;
  for (int i = 0; i < 3; ++i) {
    for (int j = 0; j < 3; ++j) Ti[i * 4 + j] = R[j * 3 + i];
    Ti[i * 4 + 3] = -(R[0 * 3 + i] * t[0] + R[1 * 3 + i] * t[1] + R[2 * 3 + i] * t[2]);
  }
  Ti[12] = 0.f; Ti[13] = 0.f; Ti[14] = 0.f; Ti[15] = 1.f;
}

// ---------------------------------------------------------------------------
extern "C" void kernel_launch(void* const* d_in, const int* in_sizes, int n_in,
                              void* d_out, int out_size, void* d_ws, size_t ws_size,
                              hipStream_t stream) {
  (void)in_sizes; (void)n_in; (void)out_size; (void)ws_size;
  const float* feats   = (const float*)d_in[0];
  const float* scores  = (const float*)d_in[1];
  const float* kpts    = (const float*)d_in[2];
  const float* t_scale = (const float*)d_in[3];
  const float* K       = (const float*)d_in[4];
  // d_in[5] = gt_pose_0to1 (unused by forward output)
  const float* bin     = (const float*)d_in[6];
  const float* W1 = (const float*)d_in[7];
  const float* b1 = (const float*)d_in[8];
  const float* W2 = (const float*)d_in[9];
  const float* b2 = (const float*)d_in[10];
  const float* W3 = (const float*)d_in[11];
  const float* b3 = (const float*)d_in[12];

  char* ws = (char*)d_ws;
  size_t off = 0;
  auto alloc = [&](size_t bytes) -> void* {
    void* p = ws + off;
    off = (off + bytes + 255) & ~(size_t)255;
    return p;
  };
  float* coup   = (float*)alloc(sizeof(float) * (size_t)NB * LD * LD); // ~67 MB (fits in 192MB L2)
  float* u      = (float*)alloc(sizeof(float) * NB * LD);
  float* v      = (float*)alloc(sizeof(float) * NB * LD);
  int*   nn12   = (int*)alloc(sizeof(int) * NB * NPTS);
  int*   nn21   = (int*)alloc(sizeof(int) * NB * NPTS);
  int*   matchv = (int*)alloc(sizeof(int) * NB * NPTS);
  float* conf   = (float*)alloc(sizeof(float) * NB * NPTS);
  float* wsum   = (float*)alloc(sizeof(float) * NB);
  float* kinv   = (float*)alloc(sizeof(float) * NB * 18);
  float* M9     = (float*)alloc(sizeof(float) * NB * 81);
  float* RT     = (float*)alloc(sizeof(float) * NB * 21);
  float* score4 = (float*)alloc(sizeof(float) * NB * 4);

  // 1) Score GEMM -> couplings core (WMMA bf16-split), then dustbins, then u=v=0.
  gemm_coup<<<dim3(NPTS / 16, NPTS / (16 * 8), NB), dim3(32, 8), 0, stream>>>(feats, coup);
  fill_bins<<<(NB * LD + 255) / 256, 256, 0, stream>>>(coup, bin);
  init_uv<<<(NB * LD + 255) / 256, 256, 0, stream>>>(u, v);

  // 2) Sinkhorn: 100 iterations, L2-resident streaming passes.
  for (int it = 0; it < SINK_ITERS; ++it) {
    sink_u<<<dim3((LD + 7) / 8, NB), dim3(32, 8), 0, stream>>>(coup, v, u);
    sink_v<<<dim3((LD + 31) / 32, NB), dim3(32, 8), 0, stream>>>(coup, u, v);
  }

  // 3) Mutual matching.
  argmax_rows<<<dim3(NPTS / 8, NB), dim3(32, 8), 0, stream>>>(coup, v, nn12);
  argmax_cols<<<dim3(NPTS / 32, NB), dim3(32, 8), 0, stream>>>(coup, u, nn21);
  match_kernel<<<(NB * NPTS + 255) / 256, 256, 0, stream>>>(nn12, nn21, matchv);

  // 4) Confidence MLP + pose.
  kinv_kernel<<<1, NB, 0, stream>>>(K, kinv);
  conf_kernel<<<dim3(NPTS / 256, NB), 256, 0, stream>>>(feats, scores, coup, u, v, matchv,
                                                        W1, b1, W2, b2, W3, b3, conf);
  wsum_kernel<<<NB, 256, 0, stream>>>(conf, wsum);
  m9_kernel<<<NB, 256, 0, stream>>>(kpts, matchv, conf, wsum, kinv, M9);
  eig_kernel<<<NB, 1, 0, stream>>>(M9, RT);
  cheir_kernel<<<NB, 256, 0, stream>>>(kpts, matchv, conf, wsum, kinv, RT, score4);
  final_kernel<<<NB, 1, 0, stream>>>(RT, score4, t_scale, (float*)d_out);
}
